// IPG_Grapher_68504728371617
// MI455X (gfx1250) — matrix operand order repaired and checked by reading, not compiled, not run on hardware
//
#include <hip/hip_runtime.h>
#include <hip/hip_bf16.h>

typedef __attribute__((ext_vector_type(16))) _Float16 v16h;
typedef __attribute__((ext_vector_type(8)))  float    v8f;
typedef __attribute__((ext_vector_type(4)))  unsigned int u32x4;
typedef __attribute__((ext_vector_type(4)))  float    f32x4;

union F16Frag { u32x4 u[2]; v16h h; };

#define DIMC   180
#define HEADS  6
#define HD     30
#define HDP    32
#define NWIN   128   // b_ = 2 * 8 * 8
#define NQ     256   // GH*GW
#define NM     576   // KH*KW
#define NPIX   32768 // 2*128*128
#define KP     192   // padded K for weight packs (K = 180 always)

// ---------------- workspace layout (bytes) ----------------
static const size_t OFF_X1    = 0;                                   // [32768,180] f32
static const size_t OFF_XS    = OFF_X1    + (size_t)NPIX*DIMC*4;     // [32768,360] f32
static const size_t OFF_X1N   = OFF_XS    + (size_t)NPIX*2*DIMC*4;   // [128,6,256,32] f16
static const size_t OFF_X2N   = OFF_X1N   + (size_t)NWIN*HEADS*NQ*HDP*2; // [128,6,576,32] f16
static const size_t OFF_FEATT = OFF_X2N   + (size_t)NWIN*HEADS*NM*HDP*2; // [128,6,32,576] f16
static const size_t OFF_ATTN  = OFF_FEATT + (size_t)NWIN*HEADS*HDP*NM*2; // [32768,180] f32
static const size_t OFF_BIAS  = OFF_ATTN  + (size_t)NPIX*DIMC*4;     // [6,256,576] f32
static const size_t OFF_TBL   = OFF_BIAS  + (size_t)HEADS*NQ*NM*4;   // [1521,6] f32
static const size_t OFF_WGP   = OFF_TBL   + 40960;                   // [192,192] f16 pack
static const size_t OFF_WSP   = OFF_WGP   + (size_t)192*KP*2;        // [384,192] f16 pack
static const size_t OFF_WPP   = OFF_WSP   + (size_t)384*KP*2;        // [192,192] f16 pack

// ---------------- CPB MLP: table[1521,2] -> relu(@W1+b1) -> @W2 -> [1521,6] ----
__global__ void cpb_kernel(const float* __restrict__ w1, const float* __restrict__ b1,
                           const float* __restrict__ w2, float* __restrict__ tbl) {
  int t = blockIdx.x * blockDim.x + threadIdx.x;
  if (t >= 1521) return;
  int i = t / 39, j = t % 39;
  float chv = (float)(i - 23) / 15.f * 8.f;
  float cwv = (float)(j - 23) / 15.f * 8.f;
  float sh = (chv > 0.f) ? 1.f : ((chv < 0.f) ? -1.f : 0.f);
  float sw = (cwv > 0.f) ? 1.f : ((cwv < 0.f) ? -1.f : 0.f);
  float a0 = sh * log2f(fabsf(chv) + 1.f) / 3.f;
  float a1 = sw * log2f(fabsf(cwv) + 1.f) / 3.f;
  float out[HEADS] = {0.f, 0.f, 0.f, 0.f, 0.f, 0.f};
  for (int k = 0; k < 512; ++k) {
    float hid = fmaxf(a0 * w1[k] + a1 * w1[512 + k] + b1[k], 0.f);
#pragma unroll
    for (int hh = 0; hh < HEADS; ++hh) out[hh] += hid * w2[k * HEADS + hh];
  }
#pragma unroll
  for (int hh = 0; hh < HEADS; ++hh) tbl[t * HEADS + hh] = out[hh];
}

// ---------------- bias16[h][n][m] = 16*sigmoid(tbl[rpi(n,m)][h]) ----------------
__global__ void bias_kernel(const float* __restrict__ tbl, float* __restrict__ bias16) {
  int t = blockIdx.x * blockDim.x + threadIdx.x;
  if (t >= NQ * NM) return;
  int n = t / NM, m = t % NM;
  int gy = n >> 4, gx = n & 15;
  int sy = m / 24, sx = m % 24;
  int idx = (sy - gy - 7) * 39 + (sx - gx - 7);
  idx %= 1521; if (idx < 0) idx += 1521;
#pragma unroll
  for (int hh = 0; hh < HEADS; ++hh) {
    float b = tbl[idx * HEADS + hh];
    bias16[hh * (NQ * NM) + t] = 16.f / (1.f + __expf(-b));
  }
}

// ------- pre-pack weight matrix B[180,N] -> Bp[Npad][192] f16, zero padded -------
__global__ void pack_w(const float* __restrict__ B, _Float16* __restrict__ P,
                       int N, int Npad) {
  int t = blockIdx.x * blockDim.x + threadIdx.x;
  if (t >= Npad * KP) return;
  int col = t / KP, kp = t % KP;
  float v = (kp < DIMC && col < N) ? B[(size_t)kp * N + col] : 0.f;
  P[(size_t)col * KP + kp] = (_Float16)v;
}

// ---------------- WMMA GEMM, K = 180 fixed: C[M,N] = A[M,180]@B + bias ----------------
// No LDS: A fragments read directly from global (fragment K-runs are contiguous
// in a row-major matrix), B fragments read from pre-packed zero-padded f16.
// block = 256 thr = 8 waves; block tile 64(M) x 32(N); fully unrolled 6 k-chunks.
__global__ __launch_bounds__(256) void gemm_f16(
    const float* __restrict__ A, const _Float16* __restrict__ Bp,
    const float* __restrict__ bias, float* __restrict__ C, int M, int N) {
  const int tid = threadIdx.x;
  const int lane = tid & 31, wave = tid >> 5;
  const int half = lane >> 4, l15 = lane & 15;
  const int wm = wave >> 1, wn = wave & 1;
  const int bm = blockIdx.x, bn = blockIdx.y;
  const int row = bm * 64 + wm * 16 + l15;
  const float* __restrict__ arow = A + (size_t)row * DIMC;
  const _Float16* __restrict__ bcol = Bp + (size_t)(bn * 32 + wn * 16 + l15) * KP;
  v8f acc = {};
#pragma unroll
  for (int kc = 0; kc < 6; ++kc) {
    F16Frag af, bf;
    const float* ar = arow + kc * 32 + half * 8;
    f32x4 a0 = *(const f32x4*)ar;
    f32x4 a1 = *(const f32x4*)(ar + 4);
#pragma unroll
    for (int i = 0; i < 4; ++i) { af.h[i] = (_Float16)a0[i]; af.h[4 + i] = (_Float16)a1[i]; }
    if (kc < 5) {
      f32x4 a2 = *(const f32x4*)(ar + 16);
      f32x4 a3 = *(const f32x4*)(ar + 20);
#pragma unroll
      for (int i = 0; i < 4; ++i) { af.h[8 + i] = (_Float16)a2[i]; af.h[12 + i] = (_Float16)a3[i]; }
    } else {
      const float* ar2 = ar + 16;   // k = 176 + half*8 + i
#pragma unroll
      for (int i = 0; i < 8; ++i) {
        int k = 176 + half * 8 + i;
        af.h[8 + i] = (k < DIMC) ? (_Float16)ar2[i] : (_Float16)0.f;
      }
    }
    bf.u[0] = *(const u32x4*)(bcol + kc * 32 + half * 8);
    bf.u[1] = *(const u32x4*)(bcol + kc * 32 + 16 + half * 8);
    acc = __builtin_amdgcn_wmma_f32_16x16x32_f16(false, af.h, false, bf.h, (short)0, acc, false, false);
  }
  const int gc = bn * 32 + wn * 16 + l15;
  if (gc < N) {
    float bv = bias[gc];
#pragma unroll
    for (int r = 0; r < 8; ++r) {
      int gr = bm * 64 + wm * 16 + half * 8 + r;
      C[(size_t)gr * N + gc] = acc[r] + bv;
    }
  }
}

// ---------------- pack + l2-normalize Q into [w][h][n][32] f16 ----------------
__global__ void pack_q(const float* __restrict__ x1, _Float16* __restrict__ x1n) {
  int t = blockIdx.x * blockDim.x + threadIdx.x;
  if (t >= NWIN * HEADS * NQ) return;
  int n = t % NQ; int h = (t / NQ) % HEADS; int w = t / (HEADS * NQ);
  int b = w >> 6, wy = (w >> 3) & 7, wx = w & 7;
  int y = wy * 16 + (n >> 4), x = wx * 16 + (n & 15);
  const float* src = x1 + ((size_t)((b * 128 + y) * 128 + x)) * DIMC + h * HD;
  float buf[HD]; float ss = 0.f;
#pragma unroll
  for (int c = 0; c < HD; ++c) { buf[c] = src[c]; ss += buf[c] * buf[c]; }
  float inv = 1.f / fmaxf(sqrtf(ss), 1e-12f);
  _Float16* dst = x1n + (size_t)t * HDP;
#pragma unroll
  for (int c = 0; c < HD; ++c) dst[c] = (_Float16)(buf[c] * inv);
  dst[30] = (_Float16)0.f; dst[31] = (_Float16)0.f;
}

// ------- pack K (normalized, [w][h][m][32]) and V transposed ([w][h][32][576]) -------
__global__ void pack_kv(const float* __restrict__ xs, _Float16* __restrict__ x2n,
                        _Float16* __restrict__ featT) {
  int t = blockIdx.x * blockDim.x + threadIdx.x;
  if (t >= NWIN * HEADS * NM) return;
  int m = t % NM; int h = (t / NM) % HEADS; int w = t / (HEADS * NM);
  int b = w >> 6, wy = (w >> 3) & 7, wx = w & 7;
  int sy = m / 24, sx = m % 24;
  int y = wy * 16 - 4 + sy, x = wx * 16 - 4 + sx;
  bool ok = (y >= 0 && y < 128 && x >= 0 && x < 128);
  float b2[HD], bfv[HD]; float ss = 0.f;
  if (ok) {
    const float* src = xs + ((size_t)((b * 128 + y) * 128 + x)) * (2 * DIMC);
#pragma unroll
    for (int c = 0; c < HD; ++c) {
      b2[c] = src[h * HD + c];
      bfv[c] = src[DIMC + h * HD + c];
      ss += b2[c] * b2[c];
    }
  } else {
#pragma unroll
    for (int c = 0; c < HD; ++c) { b2[c] = 0.f; bfv[c] = 0.f; }
  }
  float inv = 1.f / fmaxf(sqrtf(ss), 1e-12f);
  _Float16* d1 = x2n + (size_t)t * HDP;
#pragma unroll
  for (int c = 0; c < HD; ++c) d1[c] = (_Float16)(b2[c] * inv);
  d1[30] = (_Float16)0.f; d1[31] = (_Float16)0.f;
  _Float16* d2 = featT + ((size_t)(w * HEADS + h) * HDP) * NM + m;
#pragma unroll
  for (int c = 0; c < HD; ++c) d2[(size_t)c * NM] = (_Float16)bfv[c];
  d2[(size_t)30 * NM] = (_Float16)0.f; d2[(size_t)31 * NM] = (_Float16)0.f;
}

// ---------------- fused masked-cosine attention per (qtile, head, window) ----------------
__global__ __launch_bounds__(256) void attn_kernel(
    const _Float16* __restrict__ x1n,    // [128,6,256,32]
    const _Float16* __restrict__ x2n,    // [128,6,576,32]
    const _Float16* __restrict__ featT,  // [128,6,32,576]
    const unsigned char* __restrict__ graph,  // [128,6,256,576]
    const float* __restrict__ bias16,    // [6,256,576]
    const float* __restrict__ logit_scale,    // [6]
    float* __restrict__ attn_out) {           // [128*256,180]
  const int qt = blockIdx.x;   // 0..15
  const int h  = blockIdx.y;   // 0..5
  const int w  = blockIdx.z;   // 0..127

  __shared__ __align__(16) float    slab [16][584];
  __shared__ __align__(16) _Float16 probs[16][584];
  __shared__ __align__(16) float    outslab[16][32];
  __shared__ float pred[16][16];
  __shared__ float rowmax[16], rowsum[16];

  const int tid = threadIdx.x;
  const int lane = tid & 31, wave = tid >> 5;
  const int half = lane >> 4, l15 = lane & 15;
  const float scale = __expf(fminf(logit_scale[h], 4.6052f));

  for (int i = tid; i < 16 * 32; i += 256) (&outslab[0][0])[i] = 0.f;

  // ---- GEMM1: sim = Q_hat @ K_hat^T, fragments load directly from global (b128) ----
  const _Float16* qbase = x1n + ((size_t)((w * HEADS + h) * NQ) + qt * 16 + l15) * HDP;
  F16Frag af;
  af.u[0] = *(const u32x4*)(qbase + half * 8);
  af.u[1] = *(const u32x4*)(qbase + 16 + half * 8);
  for (int kt = wave; kt < 36; kt += 8) {
    const _Float16* kbase = x2n + ((size_t)((w * HEADS + h) * NM) + kt * 16 + l15) * HDP;
    F16Frag bf;
    bf.u[0] = *(const u32x4*)(kbase + half * 8);
    bf.u[1] = *(const u32x4*)(kbase + 16 + half * 8);
    v8f acc = {};
    acc = __builtin_amdgcn_wmma_f32_16x16x32_f16(false, af.h, false, bf.h, (short)0, acc, false, false);
    const int m = kt * 16 + l15;
#pragma unroll
    for (int r = 0; r < 8; ++r) {
      int q = half * 8 + r;
      int n = qt * 16 + q;
      float s = acc[r];
      bool g = graph[((size_t)((w * HEADS + h) * NQ) + n) * NM + m] != 0;
      slab[q][m] = (g ? s : -100.f) * scale + bias16[(size_t)(h * NQ + n) * NM + m];
    }
  }
  __syncthreads();

  // ---- softmax over 576 keys: 16 threads/row, unnormalized exp -> f16 probs ----
  {
    const int row = tid >> 4, seg = tid & 15;
    float mx = -1e30f;
    for (int k = seg * 36; k < seg * 36 + 36; ++k) mx = fmaxf(mx, slab[row][k]);
    pred[row][seg] = mx;
    __syncthreads();
    if (seg == 0) {
      float m2 = pred[row][0];
#pragma unroll
      for (int i = 1; i < 16; ++i) m2 = fmaxf(m2, pred[row][i]);
      rowmax[row] = m2;
    }
    __syncthreads();
    const float rm = rowmax[row];
    float ps = 0.f;
    for (int k = seg * 36; k < seg * 36 + 36; ++k) {
      float e = __expf(slab[row][k] - rm);
      probs[row][k] = (_Float16)e;
      ps += e;
    }
    pred[row][seg] = ps;
    __syncthreads();
    if (seg == 0) {
      float s2 = 0.f;
#pragma unroll
      for (int i = 0; i < 16; ++i) s2 += pred[row][i];
      rowsum[row] = s2;
    }
    __syncthreads();
  }

  // ---- GEMM2: out = P @ V (A from LDS f16 slab, B from featT, both b128) ----
  {
    const int ct = wave & 1;
    v8f acc = {};
    for (int kc = (wave >> 1); kc < 18; kc += 4) {
      const _Float16* pr = &probs[l15][kc * 32 + half * 8];
      F16Frag af2;
      af2.u[0] = *(const u32x4*)pr;
      af2.u[1] = *(const u32x4*)(pr + 16);
      const _Float16* fb = featT + ((size_t)((w * HEADS + h) * HDP) + ct * 16 + l15) * NM
                           + kc * 32 + half * 8;
      F16Frag bf2;
      bf2.u[0] = *(const u32x4*)fb;
      bf2.u[1] = *(const u32x4*)(fb + 16);
      acc = __builtin_amdgcn_wmma_f32_16x16x32_f16(false, af2.h, false, bf2.h, (short)0, acc, false, false);
    }
#pragma unroll
    for (int r = 0; r < 8; ++r)
      atomicAdd(&outslab[half * 8 + r][ct * 16 + l15], acc[r]);
  }
  __syncthreads();

  // ---- epilogue: divide by rowsum, scatter to [w*256+n][h*30+c] ----
  for (int t = tid; t < 16 * HD; t += 256) {
    int q = t / HD, c = t % HD;
    float v = outslab[q][c] / rowsum[q];
    int n = qt * 16 + q;
    attn_out[(size_t)(w * NQ + n) * DIMC + h * HD + c] = v;
  }
}

// ---------------- host-side launch ----------------
extern "C" void kernel_launch(void* const* d_in, const int* in_sizes, int n_in,
                              void* d_out, int out_size, void* d_ws, size_t ws_size,
                              hipStream_t stream) {
  const float* x_complete  = (const float*)d_in[0];
  const unsigned char* graph = (const unsigned char*)d_in[1];
  const float* w_group  = (const float*)d_in[2];
  const float* b_group  = (const float*)d_in[3];
  const float* w_sample = (const float*)d_in[4];
  const float* b_sample = (const float*)d_in[5];
  const float* w_proj   = (const float*)d_in[6];
  const float* b_proj   = (const float*)d_in[7];
  const float* cpb_w1   = (const float*)d_in[8];
  const float* cpb_b1   = (const float*)d_in[9];
  const float* cpb_w2   = (const float*)d_in[10];
  const float* lscale   = (const float*)d_in[11];

  char* ws = (char*)d_ws;
  float*     ws_x1    = (float*)(ws + OFF_X1);
  float*     ws_xs    = (float*)(ws + OFF_XS);
  _Float16*  ws_x1n   = (_Float16*)(ws + OFF_X1N);
  _Float16*  ws_x2n   = (_Float16*)(ws + OFF_X2N);
  _Float16*  ws_featT = (_Float16*)(ws + OFF_FEATT);
  float*     ws_attn  = (float*)(ws + OFF_ATTN);
  float*     ws_bias  = (float*)(ws + OFF_BIAS);
  float*     ws_tbl   = (float*)(ws + OFF_TBL);
  _Float16*  ws_wgp   = (_Float16*)(ws + OFF_WGP);
  _Float16*  ws_wsp   = (_Float16*)(ws + OFF_WSP);
  _Float16*  ws_wpp   = (_Float16*)(ws + OFF_WPP);

  // CPB bias path (independent)
  cpb_kernel<<<6, 256, 0, stream>>>(cpb_w1, cpb_b1, cpb_w2, ws_tbl);
  bias_kernel<<<(NQ * NM + 255) / 256, 256, 0, stream>>>(ws_tbl, ws_bias);

  // weight pre-packs (tiny, L2-resident)
  pack_w<<<(192 * KP + 255) / 256, 256, 0, stream>>>(w_group, ws_wgp, DIMC, 192);
  pack_w<<<(384 * KP + 255) / 256, 256, 0, stream>>>(w_sample, ws_wsp, 2 * DIMC, 384);
  pack_w<<<(192 * KP + 255) / 256, 256, 0, stream>>>(w_proj, ws_wpp, DIMC, 192);

  // input projections (WMMA GEMMs, LDS-free)
  gemm_f16<<<dim3(NPIX / 64, 192 / 32), 256, 0, stream>>>(
      x_complete, ws_wgp, b_group, ws_x1, NPIX, DIMC);
  gemm_f16<<<dim3(NPIX / 64, 384 / 32), 256, 0, stream>>>(
      x_complete, ws_wsp, b_sample, ws_xs, NPIX, 2 * DIMC);

  // normalize + pack into WMMA-fragment-friendly f16 layouts
  pack_q<<<(NWIN * HEADS * NQ + 255) / 256, 256, 0, stream>>>(ws_x1, ws_x1n);
  pack_kv<<<(NWIN * HEADS * NM + 255) / 256, 256, 0, stream>>>(ws_xs, ws_x2n, ws_featT);

  // fused masked attention
  attn_kernel<<<dim3(16, HEADS, NWIN), 256, 0, stream>>>(
      ws_x1n, ws_x2n, ws_featT, graph, ws_bias, lscale, ws_attn);

  // output projection
  gemm_f16<<<dim3(NPIX / 64, 192 / 32), 256, 0, stream>>>(
      ws_attn, ws_wpp, b_proj, (float*)d_out, NPIX, DIMC);
}
